// MultiheadAttention_7352984011416
// MI455X (gfx1250) — matrix operand order repaired and checked by reading, not compiled, not run on hardware
//
#include <hip/hip_runtime.h>
#include <hip/hip_bf16.h>
#include <math.h>

typedef __bf16 bf16_t;
typedef __attribute__((ext_vector_type(16))) __bf16 v16bf;
typedef __attribute__((ext_vector_type(8)))  __bf16 v8bf;
typedef __attribute__((ext_vector_type(4)))  __bf16 v4bf;
typedef __attribute__((ext_vector_type(2)))  __bf16 v2bf;
typedef __attribute__((ext_vector_type(8)))  float  v8f;

#define B_SZ  2
#define L_SEQ 2048
#define V_DIM 1024
#define H_N   16
#define D_H   64

static __device__ __forceinline__ v8f wmma_bf16(v16bf a, v16bf b, v8f c) {
  // v_wmma_f32_16x16x32_bf16: D = A(16x32) * B(32x16) + C, f32 accumulate
  return __builtin_amdgcn_wmma_f32_16x16x32_bf16(false, a, false, b, (short)0, c,
                                                 false, false);
}

// A operand (16x32 bf16). Layout used here (contiguous k, not AMD's table order): lane m=lane&15, hi=lane>>4;
// hi=0 holds K {0..15}, hi=1 holds K {16..31} (2 bf16/VGPR, ascending).
// Two contiguous 16B loads, concatenated (no element-wise packing).
static __device__ __forceinline__ v16bf load_a16x32(const bf16_t* t, int ld, int k0) {
  const int lane = threadIdx.x & 31;
  const int m = lane & 15, hi = lane >> 4;
  const bf16_t* p = t + m * ld + k0 + hi * 16;
  v8bf lo = *(const v8bf*)(p);
  v8bf hh = *(const v8bf*)(p + 8);
  return __builtin_shufflevector(lo, hh, 0, 1, 2, 3, 4, 5, 6, 7,
                                 8, 9, 10, 11, 12, 13, 14, 15);
}

// B operand (32x16) when the source is stored so that B's column n is a
// contiguous row n of t (i.e. t = X^T row-major, or X row-major when B=X^T).
// Lane n (hi half) reads 16 contiguous k starting at k0+hi*16: one 32B load.
static __device__ __forceinline__ v16bf load_bT(const bf16_t* t, int ld, int k0) {
  const int lane = threadIdx.x & 31;
  const int n = lane & 15, hi = lane >> 4;
  return *(const v16bf*)(t + n * ld + k0 + hi * 16);
}

// ---------------------------------------------------------------------------
// Kernel 1: per-head projections  X(4096x1024) * W_h(1024x64) -> bf16 (B,H,L,D)
// grid = (64, H, 3), block = 256 (8 waves). 64x64 output tile per block.
// W tile is staged TRANSPOSED in LDS (paired b32 stores via cvt_pk) so
// B-operands are contiguous 32B loads. Q output is pre-scaled by 1/sqrt(D).
// ---------------------------------------------------------------------------
__global__ void __launch_bounds__(256) mha_proj_kernel(
    const float* __restrict__ query, const float* __restrict__ key,
    const float* __restrict__ value, const float* __restrict__ Wq,
    const float* __restrict__ Wk, const float* __restrict__ Wv,
    bf16_t* __restrict__ Qws, bf16_t* __restrict__ Kws, bf16_t* __restrict__ Vws) {
  __shared__ bf16_t sA[64 * 32];    // 64 rows x 32 k (row-major)
  __shared__ bf16_t sBt[64 * 32];   // W^T tile: 64 d x 32 k (row-major)

  const float* X; const float* W; bf16_t* O; float oscale;
  if (blockIdx.z == 0)      { X = query; W = Wq; O = Qws; oscale = 0.125f; }
  else if (blockIdx.z == 1) { X = key;   W = Wk; O = Kws; oscale = 1.0f; }
  else                      { X = value; W = Wv; O = Vws; oscale = 1.0f; }

  const int h = blockIdx.y;
  const int grow0 = blockIdx.x * 64;
  const int lane = threadIdx.x & 31, wave = threadIdx.x >> 5;
  const int nn = lane & 15, hh = lane >> 4;
  const int orow = (wave >> 1) * 16, ocol = (wave & 1) * 32;

  // W^T staging coords: 2(k) x 4(d) chunk per thread
  const int kp = (threadIdx.x & 15) * 2;   // 0..30, even k
  const int d0 = (threadIdx.x >> 4) * 4;   // 0..60

  v8f acc0 = {}, acc1 = {};
  for (int kk = 0; kk < V_DIM; kk += 32) {
    __syncthreads();
    // stage A (64x32) as bf16 pairs (cvt_pk + b32 stores)
#pragma unroll
    for (int j = 0; j < 4; ++j) {
      int e = j * 512 + threadIdx.x * 2;     // even element index
      int rr = e >> 5, cc = e & 31;
      const float* src = X + (size_t)(grow0 + rr) * V_DIM + kk + cc;
      v2bf w; w[0] = (bf16_t)src[0]; w[1] = (bf16_t)src[1];
      *(v2bf*)(sA + e) = w;
    }
    // stage W^T (64 d x 32 k): two f32 row reads, 4 packed b32 transposed stores
    {
      const float* sp = W + ((size_t)h * V_DIM + kk + kp) * D_H + d0;
      float4 wa = *(const float4*)(sp);
      float4 wb = *(const float4*)(sp + D_H);
      v2bf w0; w0[0] = (bf16_t)wa.x; w0[1] = (bf16_t)wb.x;
      v2bf w1; w1[0] = (bf16_t)wa.y; w1[1] = (bf16_t)wb.y;
      v2bf w2; w2[0] = (bf16_t)wa.z; w2[1] = (bf16_t)wb.z;
      v2bf w3; w3[0] = (bf16_t)wa.w; w3[1] = (bf16_t)wb.w;
      *(v2bf*)(sBt + (d0 + 0) * 32 + kp) = w0;
      *(v2bf*)(sBt + (d0 + 1) * 32 + kp) = w1;
      *(v2bf*)(sBt + (d0 + 2) * 32 + kp) = w2;
      *(v2bf*)(sBt + (d0 + 3) * 32 + kp) = w3;
    }
    __syncthreads();
    v16bf a  = load_a16x32(sA + orow * 32, 32, 0);
    v16bf b0 = load_bT(sBt + ocol * 32, 32, 0);
    v16bf b1 = load_bT(sBt + (ocol + 16) * 32, 32, 0);
    acc0 = wmma_bf16(a, b0, acc0);
    acc1 = wmma_bf16(a, b1, acc1);
  }
  // C layout: VGPR r -> row r + 8*hi, col = lane&15
#pragma unroll
  for (int r = 0; r < 8; ++r) {
    int grow = grow0 + orow + r + 8 * hh;
    int bb = grow >> 11;                     // / L_SEQ
    int ll = grow & (L_SEQ - 1);
    bf16_t* dst = O + (((size_t)bb * H_N + h) * L_SEQ + ll) * D_H;
    dst[ocol + nn]      = (bf16_t)(acc0[r] * oscale);
    dst[ocol + 16 + nn] = (bf16_t)(acc1[r] * oscale);
  }
}

// ---------------------------------------------------------------------------
// Kernel 2: flash-style attention with softmax-BEFORE-mask semantics.
// grid = (L/128, B*H), block = 256 (8 waves, one 16-row query strip per wave).
// Online softmax runs over the FULL (unmasked) row; tril * value_mask gate only
// the P@V accumulation; query_mask scales the final row.
// V tile is staged TRANSPOSED in LDS (paired b32 stores) so all P@V
// B-operands are contiguous 32B loads.
// ---------------------------------------------------------------------------
__global__ void __launch_bounds__(256) mha_attn_kernel(
    const bf16_t* __restrict__ Qws, const bf16_t* __restrict__ Kws,
    const bf16_t* __restrict__ Vws, const float* __restrict__ qmask,
    const float* __restrict__ vmask, float* __restrict__ hws) {
  __shared__ bf16_t sK[32 * D_H];        // 4 KB, row-major (m, d)
  __shared__ bf16_t sVt[D_H * 32];       // 4 KB, transposed (d, m)
  __shared__ bf16_t sP[8 * 16 * 32];     // 8 KB: per-wave P transpose scratch

  const int lane = threadIdx.x & 31, wave = threadIdx.x >> 5;
  const int n = lane & 15, hi = lane >> 4;

  const int bh = blockIdx.y;             // b*H + h
  const int b  = bh / H_N;
  const int h  = bh - b * H_N;
  const int row0 = blockIdx.x * 128 + wave * 16;

  const bf16_t* Qb = Qws + (size_t)bh * L_SEQ * D_H;
  const bf16_t* Kb = Kws + (size_t)bh * L_SEQ * D_H;
  const bf16_t* Vb = Vws + (size_t)bh * L_SEQ * D_H;

  // Query strip (16 x 64, pre-scaled by 1/8) as two A operands in registers
  v16bf aq0 = load_a16x32(Qb + (size_t)row0 * D_H, D_H, 0);
  v16bf aq1 = load_a16x32(Qb + (size_t)row0 * D_H, D_H, 32);

  // V^T staging coords: 2(m) x 4(d) chunk per thread
  const int mp = (threadIdx.x & 15) * 2;   // 0..30, even m
  const int d0 = (threadIdx.x >> 4) * 4;   // 0..60

  v8f o0 = {}, o1 = {}, o2 = {}, o3 = {};
  float mst[8], lst[8];
#pragma unroll
  for (int r = 0; r < 8; ++r) { mst[r] = -1e30f; lst[r] = 0.f; }

  for (int m0 = 0; m0 < L_SEQ; m0 += 32) {
    __syncthreads();                     // protect LDS tiles from prior readers
    {  // stage K row-major (contiguous 4 KB), V transposed with paired stores
      const uint4* gk = (const uint4*)(Kb + (size_t)m0 * D_H);
      ((uint4*)sK)[threadIdx.x] = gk[threadIdx.x];
      const bf16_t* vp = Vb + (size_t)(m0 + mp) * D_H + d0;
      v4bf va = *(const v4bf*)(vp);
      v4bf vb = *(const v4bf*)(vp + D_H);
#pragma unroll
      for (int j = 0; j < 4; ++j) {
        v2bf w; w[0] = va[j]; w[1] = vb[j];
        *(v2bf*)(sVt + (d0 + j) * 32 + mp) = w;
      }
    }
    if (m0 + 32 < L_SEQ) {               // prefetch next tiles (global_prefetch_b8)
      if (threadIdx.x < 32)
        __builtin_prefetch(Kb + (size_t)(m0 + 32) * D_H + threadIdx.x * 64, 0, 1);
      else if (threadIdx.x < 64)
        __builtin_prefetch(Vb + (size_t)(m0 + 32) * D_H + (threadIdx.x - 32) * 64, 0, 1);
    }
    __syncthreads();

    // S = Q * K^T for two 16-col tiles (k = d, 64 = 2 wmma steps)
    v8f s0 = {}, s1 = {};
    {
      v16bf b00 = load_bT(sK, D_H, 0);
      v16bf b01 = load_bT(sK, D_H, 32);
      v16bf b10 = load_bT(sK + 16 * D_H, D_H, 0);
      v16bf b11 = load_bT(sK + 16 * D_H, D_H, 32);
      s0 = wmma_bf16(aq0, b00, s0); s0 = wmma_bf16(aq1, b01, s0);
      s1 = wmma_bf16(aq0, b10, s1); s1 = wmma_bf16(aq1, b11, s1);
    }

    const float vm0 = vmask[b * L_SEQ + m0 + n];
    const float vm1 = vmask[b * L_SEQ + m0 + 16 + n];
    bf16_t* pl = sP + wave * (16 * 32);

#pragma unroll
    for (int r = 0; r < 8; ++r) {
      float x0 = s0[r], x1 = s1[r];
      // full-row (unmasked) online softmax stats; row spans 16 lanes in a half
      float mx = fmaxf(x0, x1);
#pragma unroll
      for (int off = 1; off < 16; off <<= 1) mx = fmaxf(mx, __shfl_xor(mx, off, 32));
      float mnew = fmaxf(mst[r], mx);
      float corr = __expf(mst[r] - mnew);
      float p0 = __expf(x0 - mnew), p1 = __expf(x1 - mnew);
      float rs = p0 + p1;
#pragma unroll
      for (int off = 1; off < 16; off <<= 1) rs += __shfl_xor(rs, off, 32);
      lst[r] = lst[r] * corr + rs;
      mst[r] = mnew;
      o0[r] *= corr; o1[r] *= corr; o2[r] *= corr; o3[r] *= corr;

      // masked P (tril & value_mask) -> per-wave LDS scratch (C layout)
      int rowr = row0 + r + 8 * hi;
      float pm0 = (m0 + n      <= rowr) ? p0 * vm0 : 0.f;
      float pm1 = (m0 + 16 + n <= rowr) ? p1 * vm1 : 0.f;
      pl[(r + 8 * hi) * 32 + n]      = (bf16_t)pm0;
      pl[(r + 8 * hi) * 32 + 16 + n] = (bf16_t)pm1;
    }

    __builtin_amdgcn_wave_barrier();     // keep LDS store->load order
    asm volatile("" ::: "memory");

    // P (16x32) as A operand; V^T rows give contiguous B operands
    v16bf ap  = load_a16x32(pl, 32, 0);
    v16bf bv0 = load_bT(sVt + 0 * 16 * 32, 32, 0);
    v16bf bv1 = load_bT(sVt + 1 * 16 * 32, 32, 0);
    v16bf bv2 = load_bT(sVt + 2 * 16 * 32, 32, 0);
    v16bf bv3 = load_bT(sVt + 3 * 16 * 32, 32, 0);
    o0 = wmma_bf16(ap, bv0, o0);
    o1 = wmma_bf16(ap, bv1, o1);
    o2 = wmma_bf16(ap, bv2, o2);
    o3 = wmma_bf16(ap, bv3, o3);
  }

  // finalize: O / sum * query_mask, write concat layout (B, L, H, D) fp32
#pragma unroll
  for (int r = 0; r < 8; ++r) {
    int rowr = row0 + r + 8 * hi;
    float inv = qmask[b * L_SEQ + rowr] / lst[r];
    float* dst = hws + (((size_t)b * L_SEQ + rowr) * H_N + h) * D_H;
    dst[n]      = o0[r] * inv;
    dst[16 + n] = o1[r] * inv;
    dst[32 + n] = o2[r] * inv;
    dst[48 + n] = o3[r] * inv;
  }
}

// ---------------------------------------------------------------------------
// Kernel 3: y = heads_concat + query; LayerNorm over V=1024 (eps=1e-3).
// grid = B*L, block = 256.
// ---------------------------------------------------------------------------
__global__ void __launch_bounds__(256) mha_ln_kernel(
    const float* __restrict__ hws, const float* __restrict__ query,
    const float* __restrict__ gamma, const float* __restrict__ beta,
    float* __restrict__ out) {
  const int row = blockIdx.x;
  const float* hp = hws + (size_t)row * V_DIM;
  const float* qp = query + (size_t)row * V_DIM;

  float yv[4];
  float s1 = 0.f, s2 = 0.f;
#pragma unroll
  for (int j = 0; j < 4; ++j) {
    int i = threadIdx.x + j * 256;
    float y = hp[i] + qp[i];
    yv[j] = y; s1 += y; s2 += y * y;
  }
#pragma unroll
  for (int off = 16; off >= 1; off >>= 1) {
    s1 += __shfl_xor(s1, off, 32);
    s2 += __shfl_xor(s2, off, 32);
  }
  __shared__ float r1[8], r2[8];
  __shared__ float smean, srstd;
  const int wave = threadIdx.x >> 5;
  if ((threadIdx.x & 31) == 0) { r1[wave] = s1; r2[wave] = s2; }
  __syncthreads();
  if (threadIdx.x == 0) {
    float a = 0.f, c = 0.f;
#pragma unroll
    for (int w = 0; w < 8; ++w) { a += r1[w]; c += r2[w]; }
    float mean = a * (1.f / V_DIM);
    float var  = c * (1.f / V_DIM) - mean * mean;
    smean = mean; srstd = rsqrtf(var + 1e-3f);
  }
  __syncthreads();
  const float mean = smean, rstd = srstd;
#pragma unroll
  for (int j = 0; j < 4; ++j) {
    int i = threadIdx.x + j * 256;
    out[(size_t)row * V_DIM + i] = (yv[j] - mean) * rstd * gamma[i] + beta[i];
  }
}

// ---------------------------------------------------------------------------
extern "C" void kernel_launch(void* const* d_in, const int* in_sizes, int n_in,
                              void* d_out, int out_size, void* d_ws, size_t ws_size,
                              hipStream_t stream) {
  (void)in_sizes; (void)n_in; (void)out_size; (void)ws_size;
  const float* query = (const float*)d_in[0];
  const float* key   = (const float*)d_in[1];
  const float* value = (const float*)d_in[2];
  const float* qmask = (const float*)d_in[3];
  const float* vmask = (const float*)d_in[4];
  const float* Wq    = (const float*)d_in[5];
  const float* Wk    = (const float*)d_in[6];
  const float* Wv    = (const float*)d_in[7];
  const float* gamma = (const float*)d_in[8];
  const float* beta  = (const float*)d_in[9];
  float* out = (float*)d_out;

  // workspace: Q/K/V bf16 (8 MB each) + heads fp32 (16 MB) = 40 MB
  const size_t nqkv = (size_t)B_SZ * H_N * L_SEQ * D_H;   // 4 Mi elements
  bf16_t* Qws = (bf16_t*)d_ws;
  bf16_t* Kws = Qws + nqkv;
  bf16_t* Vws = Kws + nqkv;
  float*  hws = (float*)(Vws + nqkv);

  mha_proj_kernel<<<dim3((B_SZ * L_SEQ) / 64, H_N, 3), 256, 0, stream>>>(
      query, key, value, Wq, Wk, Wv, Qws, Kws, Vws);
  mha_attn_kernel<<<dim3(L_SEQ / 128, B_SZ * H_N), 256, 0, stream>>>(
      Qws, Kws, Vws, qmask, vmask, hws);
  mha_ln_kernel<<<dim3(B_SZ * L_SEQ), 256, 0, stream>>>(
      hws, query, gamma, beta, out);
}
